// TransitionBarrierNet_20529943675589
// MI455X (gfx1250) — matrix operand order repaired and checked by reading, not compiled, not run on hardware
//
#include <hip/hip_runtime.h>
#include <hip/hip_bf16.h>
#include <math.h>

#define NN 50000
#define EE 600000
#define HH 128
#define LL 6

typedef __attribute__((ext_vector_type(16))) __bf16 v16bf;
typedef __attribute__((ext_vector_type(8)))  float  v8f;

union Frag32B { uint4 u[2]; v16bf v; };

// LDS byte address of a generic pointer to __shared__ (aperture truncation, ISA 10.2).
__device__ __forceinline__ unsigned lds_addr_of(const void* p) {
    return (unsigned)(unsigned long long)p;
}

// Async memory->LDS copy of 16 bytes (GLOBAL_LOAD_ASYNC_TO_LDS_B128, ASYNCcnt).
__device__ __forceinline__ void async_copy_b128(unsigned lds_byte_addr, const void* gaddr) {
    asm volatile("global_load_async_to_lds_b128 %0, %1, off"
                 :: "v"(lds_byte_addr), "v"(gaddr)
                 : "memory");
}

__device__ __forceinline__ void async_wait0() {
    asm volatile("s_wait_asynccnt 0x0" ::: "memory");
}

// A fragment (16x32 bf16, ISA 7.12.2): lane&15 = row M; lanes 0-15 hold K 0-7 (elems 0-7)
// and K 16-23 (elems 8-15); lanes 16-31 hold K 8-15 / 24-31.
__device__ __forceinline__ v16bf load_a_frag(const __bf16* sm, int rowStride, int kt, int lane) {
    const int row  = lane & 15;
    const int base = kt * 32 + (lane >> 4) * 8;
    Frag32B f;
    f.u[0] = *(const uint4*)(sm + row * rowStride + base);
    f.u[1] = *(const uint4*)(sm + row * rowStride + base + 16);
    return f.v;
}

// B fragment: weights pre-swizzled by pack_kernel so each lane reads 32 contiguous bytes.
__device__ __forceinline__ v16bf load_b_frag(const __bf16* wp, int NT, int kt, int nt, int lane) {
    const uint4* p = (const uint4*)(wp + ((size_t)((kt * NT + nt) * 32 + lane)) * 16);
    Frag32B f;
    f.u[0] = p[0];
    f.u[1] = p[1];
    return f.v;
}

// Pack fp32 weight matrix W[K, Nc] (K%32==0, Nc%16==0) into bf16 B-fragment order:
// out[((kt*NT+nt)*32+lane)*16 + j] = W[(kt*32 + (lane>>4)*8 + (j&7) + (j>>3)*16) * Nc + nt*16 + (lane&15)]
__global__ void pack_kernel(const float* __restrict__ W, __bf16* __restrict__ out, int NT) {
    size_t idx = (size_t)blockIdx.x * 256 + threadIdx.x;
    int j    = (int)(idx & 15);
    int lane = (int)((idx >> 4) & 31);
    size_t rest = idx >> 9;
    int nt = (int)(rest % NT);
    int kt = (int)(rest / NT);
    int k = kt * 32 + (lane >> 4) * 8 + (j & 7) + (j >> 3) * 16;
    int n = nt * 16 + (lane & 15);
    out[idx] = (__bf16)W[(size_t)k * (NT * 16) + n];
}

// h = x @ node_W + node_b ; K=6 so scalar VALU. Writes fp32 + bf16 copies.
__global__ __launch_bounds__(256) void node_embed_kernel(
    const float* __restrict__ x, const float* __restrict__ W, const float* __restrict__ b,
    float* __restrict__ h, __bf16* __restrict__ hbf) {
    size_t i = (size_t)blockIdx.x * 256 + threadIdx.x; // i < N*H
    int f = (int)(i & (HH - 1));
    size_t n = i >> 7;
    float acc = b[f];
#pragma unroll
    for (int k = 0; k < 6; ++k) acc += x[n * 6 + k] * W[k * HH + f];
    h[i] = acc;
    hbf[i] = (__bf16)acc;
}

// e = [edge_attr(3) | gaussian(8)] @ edge_W + edge_b ; 2 edges per 256-thread block.
__global__ __launch_bounds__(256) void edge_embed_kernel(
    const float* __restrict__ pos, const float* __restrict__ attr,
    const int* __restrict__ eidx, const float* __restrict__ W, const float* __restrict__ b,
    __bf16* __restrict__ ebf) {
    __shared__ float feat[2][11];
    int t = threadIdx.x;
    int el = t >> 7;
    int f = t & (HH - 1);
    size_t e = (size_t)blockIdx.x * 2 + el;
    if (f == 0) {
        int s = eidx[e];
        int d = eidx[EE + e];
        float dx = pos[(size_t)d * 3 + 0] - pos[(size_t)s * 3 + 0];
        float dy = pos[(size_t)d * 3 + 1] - pos[(size_t)s * 3 + 1];
        float dz = pos[(size_t)d * 3 + 2] - pos[(size_t)s * 3 + 2];
        float len = sqrtf(dx * dx + dy * dy + dz * dz);
        feat[el][0] = attr[e * 3 + 0];
        feat[el][1] = attr[e * 3 + 1];
        feat[el][2] = attr[e * 3 + 2];
        const float width = 5.0f / 7.0f;
#pragma unroll
        for (int j = 0; j < 8; ++j) {
            float z = (len - (float)j * width) / width;
            feat[el][3 + j] = __expf(-0.5f * z * z);
        }
    }
    __syncthreads();
    float acc = b[f];
#pragma unroll
    for (int k = 0; k < 11; ++k) acc += feat[el][k] * W[k * HH + f];
    ebf[e * HH + f] = (__bf16)acc;
}

// Per-edge message MLP via WMMA: 16 edges/block, 8 waves each owning a 16-wide N tile.
// Gather rows go memory->LDS via async-copy (ASYNCcnt), no VGPR round-trip.
// m1 = [h_dst|h_src|e] @ W1 + b1 (K=384, 12 WMMA); relu; m2 = m1r @ W2 + b2 (K=128, 4 WMMA);
// scatter-add into agg[dst].
__global__ __launch_bounds__(256) void msg_kernel(
    const __bf16* __restrict__ hbf, const __bf16* __restrict__ ebf,
    const int* __restrict__ eidx,
    const __bf16* __restrict__ W1p, const float* __restrict__ b1,
    const __bf16* __restrict__ W2p, const float* __restrict__ b2,
    float* __restrict__ agg) {
    __shared__ alignas(16) __bf16 Astage[16 * 384];
    __shared__ alignas(16) __bf16 Mstage[16 * 128];

    const int t = threadIdx.x;
    const int etile = blockIdx.x;
    const int row = t >> 4;
    const int c8 = (t & 15) * 8;
    const int e = etile * 16 + row;
    const int nd = eidx[EE + e]; // dst
    const int ns = eidx[e];      // src

    {
        unsigned l0 = lds_addr_of(Astage + row * 384 + c8);
        async_copy_b128(l0,       hbf + (size_t)nd * HH + c8);
        async_copy_b128(l0 + 256, hbf + (size_t)ns * HH + c8);  // +128 bf16 cols
        async_copy_b128(l0 + 512, ebf + (size_t)e  * HH + c8);  // +256 bf16 cols
        async_wait0();
    }
    __syncthreads();

    const int w = t >> 5;
    const int lane = t & 31;
    const int khalf = lane >> 4;
    const int ncol = (w << 4) + (lane & 15);

    v8f c;
    {
        float bv = b1[ncol];
#pragma unroll
        for (int r = 0; r < 8; ++r) c[r] = bv;
    }
#pragma unroll
    for (int kt = 0; kt < 12; ++kt) {
        v16bf a = load_a_frag(Astage, 384, kt, lane);
        v16bf bb = load_b_frag(W1p, 8, kt, w, lane);
        c = __builtin_amdgcn_wmma_f32_16x16x32_bf16(false, a, false, bb, (short)0, c, false, false);
    }
#pragma unroll
    for (int r = 0; r < 8; ++r) {
        int m = r + (khalf << 3);
        Mstage[m * 128 + ncol] = (__bf16)fmaxf(c[r], 0.0f);
    }
    __syncthreads();

    v8f c2;
    {
        float bv = b2[ncol];
#pragma unroll
        for (int r = 0; r < 8; ++r) c2[r] = bv;
    }
#pragma unroll
    for (int kt = 0; kt < 4; ++kt) {
        v16bf a = load_a_frag(Mstage, 128, kt, lane);
        v16bf bb = load_b_frag(W2p, 8, kt, w, lane);
        c2 = __builtin_amdgcn_wmma_f32_16x16x32_bf16(false, a, false, bb, (short)0, c2, false, false);
    }
#pragma unroll
    for (int r = 0; r < 8; ++r) {
        int m = r + (khalf << 3);
        int node = eidx[EE + etile * 16 + m];
        unsafeAtomicAdd(&agg[(size_t)node * HH + ncol], c2[r]);
    }
}

// Node update MLP + residual + LayerNorm via WMMA: 16 nodes/block, 8 waves.
__global__ __launch_bounds__(256) void upd_kernel(
    float* __restrict__ h, __bf16* __restrict__ hbf,
    const float* __restrict__ agg,
    const __bf16* __restrict__ U1p, const float* __restrict__ b1,
    const __bf16* __restrict__ U2p, const float* __restrict__ b2,
    const float* __restrict__ lng, const float* __restrict__ lnb) {
    __shared__ alignas(16) __bf16 Astage[16 * 256];
    __shared__ alignas(16) __bf16 Ustage[16 * 128];
    __shared__ alignas(16) float  Fstage[16 * 128];
    __shared__ float mu_s[16], rs_s[16];

    const int t = threadIdx.x;
    const int ntile = blockIdx.x;
    const int row = t >> 4;
    const int c8 = (t & 15) * 8;
    const size_t node = (size_t)ntile * 16 + row;

    // h rows: async memory->LDS; agg rows: fp32->bf16 convert through VGPRs.
    async_copy_b128(lds_addr_of(Astage + row * 256 + c8), hbf + node * HH + c8);
    {
        const float4* ap = (const float4*)(agg + node * HH + c8);
        float4 a0 = ap[0], a1 = ap[1];
        __bf16* d = Astage + row * 256 + 128 + c8;
        d[0] = (__bf16)a0.x; d[1] = (__bf16)a0.y; d[2] = (__bf16)a0.z; d[3] = (__bf16)a0.w;
        d[4] = (__bf16)a1.x; d[5] = (__bf16)a1.y; d[6] = (__bf16)a1.z; d[7] = (__bf16)a1.w;
    }
    async_wait0();
    __syncthreads();

    const int w = t >> 5;
    const int lane = t & 31;
    const int khalf = lane >> 4;
    const int ncol = (w << 4) + (lane & 15);

    v8f c;
    {
        float bv = b1[ncol];
#pragma unroll
        for (int r = 0; r < 8; ++r) c[r] = bv;
    }
#pragma unroll
    for (int kt = 0; kt < 8; ++kt) {
        v16bf a = load_a_frag(Astage, 256, kt, lane);
        v16bf bb = load_b_frag(U1p, 8, kt, w, lane);
        c = __builtin_amdgcn_wmma_f32_16x16x32_bf16(false, a, false, bb, (short)0, c, false, false);
    }
#pragma unroll
    for (int r = 0; r < 8; ++r) {
        int m = r + (khalf << 3);
        Ustage[m * 128 + ncol] = (__bf16)fmaxf(c[r], 0.0f);
    }
    __syncthreads();

    v8f c2;
    {
        float bv = b2[ncol];
#pragma unroll
        for (int r = 0; r < 8; ++r) c2[r] = bv;
    }
#pragma unroll
    for (int kt = 0; kt < 4; ++kt) {
        v16bf a = load_a_frag(Ustage, 128, kt, lane);
        v16bf bb = load_b_frag(U2p, 8, kt, w, lane);
        c2 = __builtin_amdgcn_wmma_f32_16x16x32_bf16(false, a, false, bb, (short)0, c2, false, false);
    }
    // residual add, stash fp32 for LayerNorm
#pragma unroll
    for (int r = 0; r < 8; ++r) {
        int m = r + (khalf << 3);
        size_t nd = (size_t)ntile * 16 + m;
        Fstage[m * 128 + ncol] = c2[r] + h[nd * HH + ncol];
    }
    __syncthreads();
    if (t < 16) {
        float s = 0.0f;
        for (int j = 0; j < 128; ++j) s += Fstage[t * 128 + j];
        float mu = s * (1.0f / 128.0f);
        float v = 0.0f;
        for (int j = 0; j < 128; ++j) {
            float d = Fstage[t * 128 + j] - mu;
            v += d * d;
        }
        v *= (1.0f / 128.0f);
        mu_s[t] = mu;
        rs_s[t] = rsqrtf(v + 1e-5f);
    }
    __syncthreads();
    {
        float mu = mu_s[row], rs = rs_s[row];
#pragma unroll
        for (int j = 0; j < 8; ++j) {
            int f = c8 + j;
            float val = (Fstage[row * 128 + f] - mu) * rs * lng[f] + lnb[f];
            h[node * HH + f] = val;
            hbf[node * HH + f] = (__bf16)val;
        }
    }
}

// Mean-pool per graph id via atomics.
__global__ __launch_bounds__(256) void pool_kernel(
    const float* __restrict__ h, const int* __restrict__ batch,
    float* __restrict__ pool, float* __restrict__ cnt) {
    size_t i = (size_t)blockIdx.x * 256 + threadIdx.x;
    int f = (int)(i & (HH - 1));
    size_t n = i >> 7;
    int bb = batch[n];
    unsafeAtomicAdd(&pool[(size_t)bb * HH + f], h[i]);
    if (f == 0) unsafeAtomicAdd(&cnt[bb], 1.0f);
}

// Tiny head MLP, single block.
__global__ __launch_bounds__(256) void head_kernel(
    const float* __restrict__ pool, const float* __restrict__ cnt,
    const float* __restrict__ W1, const float* __restrict__ b1,
    const float* __restrict__ W2, const float* __restrict__ b2,
    const float* __restrict__ W3, const float* __restrict__ b3,
    float* __restrict__ out) {
    __shared__ float comb[256], z1[128], z2[64];
    int t = threadIdx.x;
    int g = t >> 7, f = t & 127;
    comb[t] = pool[g * HH + f] / fmaxf(cnt[g], 1.0f);
    __syncthreads();
    if (t < 128) {
        float a = b1[t];
        for (int k = 0; k < 256; ++k) a += comb[k] * W1[k * 128 + t];
        z1[t] = fmaxf(a, 0.0f);
    }
    __syncthreads();
    if (t < 64) {
        float a = b2[t];
        for (int k = 0; k < 128; ++k) a += z1[k] * W2[k * 64 + t];
        z2[t] = fmaxf(a, 0.0f);
    }
    __syncthreads();
    if (t == 0) {
        float a = b3[0];
        for (int k = 0; k < 64; ++k) a += z2[k] * W3[k];
        out[0] = (a > 20.0f) ? a : log1pf(__expf(a));
    }
}

extern "C" void kernel_launch(void* const* d_in, const int* in_sizes, int n_in,
                              void* d_out, int out_size, void* d_ws, size_t ws_size,
                              hipStream_t stream) {
    const float* x        = (const float*)d_in[0];
    const float* pos      = (const float*)d_in[1];
    const float* eattr    = (const float*)d_in[2];
    const int*   eidx     = (const int*)d_in[3];
    const int*   batch    = (const int*)d_in[4];
    const float* node_W   = (const float*)d_in[5];
    const float* node_b   = (const float*)d_in[6];
    const float* edge_W   = (const float*)d_in[7];
    const float* edge_b   = (const float*)d_in[8];
    const float* msgW1    = (const float*)d_in[9];
    const float* msgb1    = (const float*)d_in[10];
    const float* msgW2    = (const float*)d_in[11];
    const float* msgb2    = (const float*)d_in[12];
    const float* updW1    = (const float*)d_in[13];
    const float* updb1    = (const float*)d_in[14];
    const float* updW2    = (const float*)d_in[15];
    const float* updb2    = (const float*)d_in[16];
    const float* ln_g     = (const float*)d_in[17];
    const float* ln_b     = (const float*)d_in[18];
    const float* headW1   = (const float*)d_in[19];
    const float* headb1   = (const float*)d_in[20];
    const float* headW2   = (const float*)d_in[21];
    const float* headb2   = (const float*)d_in[22];
    const float* headW3   = (const float*)d_in[23];
    const float* headb3   = (const float*)d_in[24];
    float* out = (float*)d_out;

    // Workspace carve-up (256B aligned)
    char* ws = (char*)d_ws;
    size_t off = 0;
    auto take = [&](size_t bytes) {
        size_t o = off;
        off = (off + bytes + 255) & ~(size_t)255;
        return o;
    };
    float*  h_f   = (float*)(ws + take((size_t)NN * HH * 4));
    __bf16* h_bf  = (__bf16*)(ws + take((size_t)NN * HH * 2));
    float*  agg   = (float*)(ws + take((size_t)NN * HH * 4));
    __bf16* e_bf  = (__bf16*)(ws + take((size_t)EE * HH * 2));
    __bf16* W1p   = (__bf16*)(ws + take((size_t)LL * 3 * HH * HH * 2));
    __bf16* W2p   = (__bf16*)(ws + take((size_t)LL * HH * HH * 2));
    __bf16* U1p   = (__bf16*)(ws + take((size_t)LL * 2 * HH * HH * 2));
    __bf16* U2p   = (__bf16*)(ws + take((size_t)LL * HH * HH * 2));
    float*  pool  = (float*)(ws + take(2 * HH * 4));
    float*  cnt   = (float*)(ws + take(2 * 4));
    (void)ws_size; (void)in_sizes; (void)n_in; (void)out_size;

    // 1) Pack weights to bf16 B-fragment layout (tiny, once per call)
    for (int l = 0; l < LL; ++l) {
        pack_kernel<<<(3 * HH * HH) / 256, 256, 0, stream>>>(msgW1 + (size_t)l * 3 * HH * HH, W1p + (size_t)l * 3 * HH * HH, HH / 16);
        pack_kernel<<<(HH * HH) / 256, 256, 0, stream>>>(msgW2 + (size_t)l * HH * HH, W2p + (size_t)l * HH * HH, HH / 16);
        pack_kernel<<<(2 * HH * HH) / 256, 256, 0, stream>>>(updW1 + (size_t)l * 2 * HH * HH, U1p + (size_t)l * 2 * HH * HH, HH / 16);
        pack_kernel<<<(HH * HH) / 256, 256, 0, stream>>>(updW2 + (size_t)l * HH * HH, U2p + (size_t)l * HH * HH, HH / 16);
    }

    // 2) Embeddings
    node_embed_kernel<<<(NN * HH) / 256, 256, 0, stream>>>(x, node_W, node_b, h_f, h_bf);
    edge_embed_kernel<<<EE / 2, 256, 0, stream>>>(pos, eattr, eidx, edge_W, edge_b, e_bf);

    // 3) Message-passing layers
    for (int l = 0; l < LL; ++l) {
        hipMemsetAsync(agg, 0, (size_t)NN * HH * 4, stream);
        msg_kernel<<<EE / 16, 256, 0, stream>>>(
            h_bf, e_bf, eidx,
            W1p + (size_t)l * 3 * HH * HH, msgb1 + (size_t)l * HH,
            W2p + (size_t)l * HH * HH, msgb2 + (size_t)l * HH,
            agg);
        upd_kernel<<<NN / 16, 256, 0, stream>>>(
            h_f, h_bf, agg,
            U1p + (size_t)l * 2 * HH * HH, updb1 + (size_t)l * HH,
            U2p + (size_t)l * HH * HH, updb2 + (size_t)l * HH,
            ln_g + (size_t)l * HH, ln_b + (size_t)l * HH);
    }

    // 4) Pool + head
    hipMemsetAsync(pool, 0, 2 * HH * 4, stream);
    hipMemsetAsync(cnt, 0, 2 * 4, stream);
    pool_kernel<<<(NN * HH) / 256, 256, 0, stream>>>(h_f, batch, pool, cnt);
    head_kernel<<<1, 256, 0, stream>>>(pool, cnt, headW1, headb1, headW2, headb2, headW3, headb3, out);
}
